// MLP_4887672783348
// MI455X (gfx1250) — compile-verified, matrix operand order checked by reading, not run on hardware
//
#include <hip/hip_runtime.h>
#include <hip/hip_bf16.h>
#include <stdint.h>

// ---------------------------------------------------------------------------
// Fused 23-layer MLP for MI455X (gfx1250, wave32, WMMA).
//   x:[32768,3] f32 -> 2048 -> 2048 -> ... -> 1 (ReLU between layers)
// Strategy:
//   * Pre-convert all weights f32->f16 into d_ws (L2-resident, ~17MB).
//   * One block = 32 batch rows. Activations ping-pong in LDS as f16
//     (rows padded +8 halves for bank-conflict-free column access).
//   * WMMA layers: each wave computes BOTH m-tiles for its n-tile(s), so
//     every weight fragment loaded from L2 feeds 2 WMMAs (min weight
//     traffic = 1x W per workgroup). Big layers additionally pair two
//     n-tiles per wave so each A fragment feeds 2 WMMAs.
//   * Per-row fragment base pointers are hoisted out of the K loop so all
//     ds/global loads use immediate offsets (no per-iteration VALU adds).
//   * Tiny head (3->2048) and tail (<=16 wide) layers run on VALU.
// ---------------------------------------------------------------------------

#define NL 23
#define DIMS_INIT {3,2048,2048,1024,1024,512,512,256,256,128,128,64,64,32,32,16,16,8,8,4,4,2,2,1}
#define MROWS 32        // batch rows per workgroup
#define LDPAD 8         // halves of row padding in LDS (stride 4112B -> bank-friendly)
#define MAXW 2048       // widest activation
#define ROWSTRIDE (MAXW + LDPAD)   // 2056 halves per LDS row slot

static const int h_dims[NL + 1] = DIMS_INIT;
__device__ __constant__ int c_dims[NL + 1] = DIMS_INIT;

typedef __attribute__((ext_vector_type(16))) _Float16 v16h;
typedef __attribute__((ext_vector_type(8)))  _Float16 v8h;
typedef __attribute__((ext_vector_type(8)))  float    v8f;

struct MLPParams {
    const float* W[NL];   // fp32 weights (used by layer 0 only)
    const float* b[NL];   // fp32 biases
};

// ---------------------------------------------------------------------------
// Weight conversion: f32 -> f16 into workspace.
// ---------------------------------------------------------------------------
__global__ void cvt_f32_to_f16(const float* __restrict__ src,
                               _Float16* __restrict__ dst, int n) {
    int i = blockIdx.x * blockDim.x + threadIdx.x;
    if (i < n) dst[i] = (_Float16)src[i];
}

// Load a 16-halves fragment as two 16B chunks (ds_load_b128/global_load_b128).
__device__ __forceinline__ v16h ldfrag(const _Float16* p) {
    v8h lo = *(const v8h*)p;
    v8h hi = *(const v8h*)(p + 16);
    return __builtin_shufflevector(lo, hi,
        0,1,2,3,4,5,6,7,8,9,10,11,12,13,14,15);
}

// Bias + ReLU + store one 16x16 f32 accumulator tile as f16 into LDS.
__device__ __forceinline__ void st_tile(const v8f& acc, float bias,
                                        _Float16* pout, int m0, int n, int hsel) {
#pragma unroll
    for (int r = 0; r < 8; ++r) {
        const int m = m0 + r + (hsel << 3);
        float v = acc[r] + bias;
        v = v > 0.f ? v : 0.f;
        pout[(size_t)m * ROWSTRIDE + n] = (_Float16)v;
    }
}

// ---------------------------------------------------------------------------
// Fused MLP kernel. grid = 32768/MROWS blocks, block = 256 threads (8 waves).
// Dynamic LDS: bufA | bufB (MROWS*ROWSTRIDE halves each) | xs (MROWS*3 f32)
// ---------------------------------------------------------------------------
__global__ void __launch_bounds__(256)
mlp_fused(const float* __restrict__ x, MLPParams P,
          const _Float16* __restrict__ wh, float* __restrict__ out) {
    extern __shared__ _Float16 smem[];
    _Float16* bufA = smem;
    _Float16* bufB = smem + (size_t)MROWS * ROWSTRIDE;
    float*    xs   = (float*)(smem + (size_t)2 * MROWS * ROWSTRIDE);

    const int tid    = threadIdx.x;
    const int lane   = tid & 31;
    const int wave   = tid >> 5;
    const int lane16 = lane & 15;
    const int hsel   = lane >> 4;           // which K-half this lane holds
    const int r0     = blockIdx.x * MROWS;  // first batch row of this block

    // ---------------- Layer 0: 3 -> 2048 (VALU) -------------------------
    for (int i = tid; i < MROWS * 3; i += 256)
        xs[i] = x[(size_t)(r0 + i / 3) * 3 + (i % 3)];
    __syncthreads();

    {
        const float* W0 = P.W[0];
        const float* b0 = P.b[0];
        for (int idx = tid; idx < MROWS * 2048; idx += 256) {
            const int m = idx >> 11;
            const int n = idx & 2047;
            float a = b0[n]
                    + xs[m * 3 + 0] * W0[n * 3 + 0]
                    + xs[m * 3 + 1] * W0[n * 3 + 1]
                    + xs[m * 3 + 2] * W0[n * 3 + 2];
            bufA[(size_t)m * ROWSTRIDE + n] = (_Float16)(a > 0.f ? a : 0.f);
        }
    }
    __syncthreads();

    size_t    woff = (size_t)2048 * 3;  // f16 weight offset past layer 0
    _Float16* pin  = bufA;
    _Float16* pout = bufB;

    // ---------------- Layers 1..14: WMMA f16 ----------------------------
    for (int l = 1; l <= 14; ++l) {
        const int K = c_dims[l];
        const int N = c_dims[l + 1];
        const float* pb = P.b[l];
        const int Nt = N >> 4;  // number of 16-wide n tiles

        // Per-lane base pointers for the two A m-tiles (loop-invariant rows).
        const _Float16* aRow0 = pin + (size_t)lane16        * ROWSTRIDE + (hsel << 3);
        const _Float16* aRow1 = pin + (size_t)(16 + lane16) * ROWSTRIDE + (hsel << 3);

        if (Nt >= 16) {
            // Big layers: 2 m-tiles x 2 n-tiles per wave iteration.
            // Per K step: 4 ds_b128 + 4 global_b128 -> 4 WMMAs.
            for (int nt = wave * 2; nt < Nt; nt += 16) {
                const int n0 = nt << 4;
                const int n1 = n0 + 16;
                const _Float16* bRow0 =
                    wh + woff + (size_t)(n0 + lane16) * K + (hsel << 3);
                const _Float16* bRow1 =
                    wh + woff + (size_t)(n1 + lane16) * K + (hsel << 3);

                v8f acc00 = {}, acc10 = {}, acc01 = {}, acc11 = {};

                for (int kk = 0; kk < K; kk += 32) {
                    v16h a0 = ldfrag(aRow0 + kk);
                    v16h a1 = ldfrag(aRow1 + kk);
                    v16h b0 = ldfrag(bRow0 + kk);
                    v16h b1 = ldfrag(bRow1 + kk);

                    acc00 = __builtin_amdgcn_wmma_f32_16x16x32_f16(
                        false, a0, false, b0, (short)0, acc00, false, false);
                    acc10 = __builtin_amdgcn_wmma_f32_16x16x32_f16(
                        false, a1, false, b0, (short)0, acc10, false, false);
                    acc01 = __builtin_amdgcn_wmma_f32_16x16x32_f16(
                        false, a0, false, b1, (short)0, acc01, false, false);
                    acc11 = __builtin_amdgcn_wmma_f32_16x16x32_f16(
                        false, a1, false, b1, (short)0, acc11, false, false);
                }

                const float bias0 = pb[n0 + lane16];
                const float bias1 = pb[n1 + lane16];
                st_tile(acc00, bias0, pout,  0, n0 + lane16, hsel);
                st_tile(acc10, bias0, pout, 16, n0 + lane16, hsel);
                st_tile(acc01, bias1, pout,  0, n1 + lane16, hsel);
                st_tile(acc11, bias1, pout, 16, n1 + lane16, hsel);
            }
        } else {
            // Narrow layers: 2 m-tiles x 1 n-tile per wave iteration.
            for (int nt = wave; nt < Nt; nt += 8) {
                const int n0 = nt << 4;
                const _Float16* bRow0 =
                    wh + woff + (size_t)(n0 + lane16) * K + (hsel << 3);

                v8f acc00 = {}, acc10 = {};

                for (int kk = 0; kk < K; kk += 32) {
                    v16h a0 = ldfrag(aRow0 + kk);
                    v16h a1 = ldfrag(aRow1 + kk);
                    v16h b0 = ldfrag(bRow0 + kk);

                    acc00 = __builtin_amdgcn_wmma_f32_16x16x32_f16(
                        false, a0, false, b0, (short)0, acc00, false, false);
                    acc10 = __builtin_amdgcn_wmma_f32_16x16x32_f16(
                        false, a1, false, b0, (short)0, acc10, false, false);
                }

                const float bias0 = pb[n0 + lane16];
                st_tile(acc00, bias0, pout,  0, n0 + lane16, hsel);
                st_tile(acc10, bias0, pout, 16, n0 + lane16, hsel);
            }
        }

        woff += (size_t)N * K;
        __syncthreads();
        _Float16* tmp = pin; pin = pout; pout = tmp;
    }

    // ---------------- Layers 15..22: tiny tail (VALU) -------------------
    for (int l = 15; l < NL; ++l) {
        const int K = c_dims[l];
        const int N = c_dims[l + 1];
        if (tid < MROWS) {
            const int m = tid;
            const float* pb = P.b[l];
            for (int n = 0; n < N; ++n) {
                float acc = pb[n];
                const _Float16* wr = wh + woff + (size_t)n * K;
                for (int k = 0; k < K; ++k)
                    acc += (float)pin[(size_t)m * ROWSTRIDE + k] * (float)wr[k];
                if (l == NL - 1) {
                    out[r0 + m] = acc;              // last layer: no ReLU, f32
                } else {
                    pout[(size_t)m * ROWSTRIDE + n] =
                        (_Float16)(acc > 0.f ? acc : 0.f);
                }
            }
        }
        woff += (size_t)N * K;
        __syncthreads();
        _Float16* tmp = pin; pin = pout; pout = tmp;
    }
}

// ---------------------------------------------------------------------------
// Host launcher. d_in order: x, W0, b0, W1, b1, ..., W22, b22.
// ---------------------------------------------------------------------------
extern "C" void kernel_launch(void* const* d_in, const int* in_sizes, int n_in,
                              void* d_out, int out_size, void* d_ws, size_t ws_size,
                              hipStream_t stream) {
    (void)in_sizes; (void)n_in; (void)out_size; (void)ws_size;

    const float* x = (const float*)d_in[0];
    MLPParams P;
    for (int l = 0; l < NL; ++l) {
        P.W[l] = (const float*)d_in[1 + 2 * l];
        P.b[l] = (const float*)d_in[2 + 2 * l];
    }

    _Float16* wh = (_Float16*)d_ws;

    // Convert all weights f32 -> f16 into workspace (~16.8 MB of halves).
    size_t off = 0;
    for (int l = 0; l < NL; ++l) {
        const int n = h_dims[l] * h_dims[l + 1];
        const int blocks = (n + 255) / 256;
        cvt_f32_to_f16<<<blocks, 256, 0, stream>>>(P.W[l], wh + off, n);
        off += (size_t)n;
    }

    const int batch  = 32768;
    const int blocks = batch / MROWS;                       // 1024
    const size_t lds = (size_t)2 * MROWS * ROWSTRIDE * sizeof(_Float16)
                     + (size_t)MROWS * 3 * sizeof(float);   // ~263.5 KB < 320 KB

    mlp_fused<<<dim3(blocks), dim3(256), lds, stream>>>(x, P, wh, (float*)d_out);
}
